// GCNConvModel_59399397704019
// MI455X (gfx1250) — matrix operand order repaired and checked by reading, not compiled, not run on hardware
//
#include <hip/hip_runtime.h>

typedef __attribute__((ext_vector_type(16))) _Float16 v16h;
typedef __attribute__((ext_vector_type(8)))  float    v8f;

#define IN_C  64
#define HID   128
#define OUT_C 64

// ---------------------------------------------------------------------------
// Kernel 1: deg[n] = 1.0f  (self-loop contribution to in-degree)
// ---------------------------------------------------------------------------
__global__ __launch_bounds__(256) void init_deg_kernel(float* __restrict__ deg, int n) {
    int i = blockIdx.x * 256 + threadIdx.x;
    if (i < n) deg[i] = 1.0f;
}

// ---------------------------------------------------------------------------
// Kernel 2: deg[dst[e]] += 1 over all real edges
// ---------------------------------------------------------------------------
__global__ __launch_bounds__(256) void deg_kernel(const int* __restrict__ dst,
                                                  float* __restrict__ deg, int e_cnt) {
    int e = blockIdx.x * 256 + threadIdx.x;
    if (e < e_cnt) atomicAdd(&deg[dst[e]], 1.0f);
}

// ---------------------------------------------------------------------------
// Kernel 3: dinv[n] = rsqrt(deg[n])   (deg >= 1 always due to self loop)
// ---------------------------------------------------------------------------
__global__ __launch_bounds__(256) void dinv_kernel(float* __restrict__ deg, int n) {
    int i = blockIdx.x * 256 + threadIdx.x;
    if (i < n) deg[i] = rsqrtf(deg[i]);
}

// ---------------------------------------------------------------------------
// Kernel 4: fused  h2 = relu(X@W1 + b1) @ W2   via v_wmma_f32_16x16x32_f16.
// One wave per 16-node tile. Weights live transposed in LDS so each B operand
// is 16 contiguous halves per lane. GEMM1 results staged (f16) in a per-wave
// LDS buffer to convert C-layout -> A-layout for GEMM2.
// ---------------------------------------------------------------------------
__global__ __launch_bounds__(256) void mlp_wmma_kernel(
    const float* __restrict__ X,    // [N, 64]
    const float* __restrict__ W1,   // [64, 128] row-major
    const float* __restrict__ B1,   // [128]
    const float* __restrict__ W2,   // [128, 64] row-major
    float* __restrict__ H2,         // [N, 64] out
    int nTiles)
{
    __shared__ __align__(32) _Float16 sW1[HID * IN_C];   // transposed: [n][k], 16 KB
    __shared__ __align__(32) _Float16 sW2[OUT_C * HID];  // transposed: [n][k], 16 KB
    __shared__ float sB1[HID];
    __shared__ __align__(32) _Float16 sH[8][16 * HID];   // per-wave staging, 32 KB

    const int tid = threadIdx.x;

    // Cooperative load + transpose of weights into LDS (f32 -> f16)
    for (int idx = tid; idx < IN_C * HID; idx += 256) {
        int k = idx >> 7, n = idx & 127;                 // W1[k][n], coalesced read
        sW1[n * IN_C + k] = (_Float16)W1[idx];
    }
    for (int idx = tid; idx < HID * OUT_C; idx += 256) {
        int k = idx >> 6, n = idx & 63;                  // W2[k][n], coalesced read
        sW2[n * HID + k] = (_Float16)W2[idx];
    }
    if (tid < HID) sB1[tid] = B1[tid];
    __syncthreads();

    const int lane   = tid & 31;
    const int laneLo = lane & 15;    // C/D column, A row
    const int hi     = lane >> 4;    // lane-half selector
    const int wslot  = tid >> 5;
    _Float16* hrow = &sH[wslot][0];

    const int wave = blockIdx.x * 8 + wslot;
    const int nw   = gridDim.x * 8;

    for (int t = wave; t < nTiles; t += nw) {
        const float* xr = X + (long long)(t * 16 + laneLo) * IN_C;

        // --- A1: 16x32 f16 tiles from X (contiguous k order, k = 16*hi + s) -----------------
        // lane<16: halves 0..15 -> K g..g+15 (g=0)
        // lane>=16: same with g=16
        v16h a1[2];
#pragma unroll
        for (int kb = 0; kb < 2; ++kb) {
            const int base = kb * 32 + hi * 16;
#pragma unroll
            for (int i = 0; i < 8; ++i) {
                a1[kb][i]     = (_Float16)xr[base + i];
                a1[kb][i + 8] = (_Float16)xr[base + 8 + i];
            }
        }

        // --- GEMM1: 16x128 = 16x64 @ 64x128, + bias + relu -> sH -------
#pragma unroll
        for (int nb = 0; nb < 8; ++nb) {
            v8f acc = {};
#pragma unroll
            for (int kb = 0; kb < 2; ++kb) {
                // B operand: Wt[N][Kgroup + i], 16 contiguous halves (32B aligned)
                const v16h b = *(const v16h*)&sW1[(nb * 16 + laneLo) * IN_C + kb * 32 + hi * 16];
                acc = __builtin_amdgcn_wmma_f32_16x16x32_f16(
                    false, a1[kb], false, b, (short)0, acc, false, false);
            }
            const float bias = sB1[nb * 16 + laneLo];
#pragma unroll
            for (int r = 0; r < 8; ++r) {
                float v = acc[r] + bias;
                v = v > 0.f ? v : 0.f;
                // C layout: M = hi*8 + r, N = nb*16 + laneLo
                hrow[(hi * 8 + r) * HID + nb * 16 + laneLo] = (_Float16)v;
            }
        }

        // --- GEMM2: 16x64 = 16x128 @ 128x64 ----------------------------
#pragma unroll
        for (int nb = 0; nb < 4; ++nb) {
            v8f acc = {};
#pragma unroll
            for (int kb = 0; kb < 4; ++kb) {
                v16h a;
                const _Float16* p0 = &hrow[laneLo * HID + kb * 32 + hi * 16];
#pragma unroll
                for (int i = 0; i < 8; ++i) { a[i] = p0[i]; a[i + 8] = p0[8 + i]; }
                const v16h b = *(const v16h*)&sW2[(nb * 16 + laneLo) * HID + kb * 32 + hi * 16];
                acc = __builtin_amdgcn_wmma_f32_16x16x32_f16(
                    false, a, false, b, (short)0, acc, false, false);
            }
#pragma unroll
            for (int r = 0; r < 8; ++r) {
                H2[(long long)(t * 16 + hi * 8 + r) * OUT_C + nb * 16 + laneLo] = acc[r];
            }
        }
    }
}

// ---------------------------------------------------------------------------
// Kernel 5: out[n][c] = b2[c] + h2[n][c] * dinv[n]^2   (self-loop message)
// ---------------------------------------------------------------------------
__global__ __launch_bounds__(256) void init_out_kernel(
    const float* __restrict__ h2, const float* __restrict__ dinv,
    const float* __restrict__ b2, float* __restrict__ out, int n)
{
    long long idx = (long long)blockIdx.x * 256 + threadIdx.x;
    if (idx >= (long long)n * OUT_C) return;
    int node = (int)(idx >> 6);
    int c    = (int)(idx & 63);
    float di = dinv[node];
    out[idx] = b2[c] + h2[idx] * di * di;
}

// ---------------------------------------------------------------------------
// Kernel 6: edge scatter. 16 lanes per edge, float4 per lane (64 channels).
// Fully coalesced 256B gather of h2[src], global_atomic_add_f32 scatter.
// ---------------------------------------------------------------------------
__global__ __launch_bounds__(256) void scatter_kernel(
    const int* __restrict__ src, const int* __restrict__ dst,
    const float* __restrict__ dinv, const float* __restrict__ h2,
    float* __restrict__ out, int e_cnt)
{
    long long tid = (long long)blockIdx.x * 256 + threadIdx.x;
    int e   = (int)(tid >> 4);
    int sub = (int)(tid & 15);
    if (e >= e_cnt) return;
    int s = src[e], d = dst[e];
    float norm = dinv[s] * dinv[d];
    const float4 v = *(const float4*)(h2 + (long long)s * OUT_C + sub * 4);
    float* op = out + (long long)d * OUT_C + sub * 4;
    atomicAdd(op + 0, v.x * norm);
    atomicAdd(op + 1, v.y * norm);
    atomicAdd(op + 2, v.z * norm);
    atomicAdd(op + 3, v.w * norm);
}

// ---------------------------------------------------------------------------
extern "C" void kernel_launch(void* const* d_in, const int* in_sizes, int n_in,
                              void* d_out, int out_size, void* d_ws, size_t ws_size,
                              hipStream_t stream)
{
    const float* X  = (const float*)d_in[0];   // [N, 64]
    const int*   EI = (const int*)d_in[1];     // [2, E] int32 flat
    const float* W1 = (const float*)d_in[2];   // [64, 128]
    const float* B1 = (const float*)d_in[3];   // [128]
    const float* W2 = (const float*)d_in[4];   // [128, 64]
    const float* B2 = (const float*)d_in[5];   // [64]
    float* out = (float*)d_out;                // [N, 64]

    const int N_ = in_sizes[0] / IN_C;         // 100000
    const int E_ = in_sizes[1] / 2;            // 1600000
    const int* src = EI;
    const int* dst = EI + E_;

    // workspace: deg/dinv [N] floats, then h2 [N*64] floats
    float* deg = (float*)d_ws;
    float* h2  = deg + N_;

    // 1) deg = 1 (self loops)
    init_deg_kernel<<<(N_ + 255) / 256, 256, 0, stream>>>(deg, N_);
    // 2) in-degree atomics over edges
    deg_kernel<<<(E_ + 255) / 256, 256, 0, stream>>>(dst, deg, E_);
    // 3) dinv = rsqrt(deg) in place
    dinv_kernel<<<(N_ + 255) / 256, 256, 0, stream>>>(deg, N_);
    // 4) fused fc1+relu+fc2 via WMMA -> h2  (N is an exact multiple of 16)
    const int nTiles = N_ / 16;
    int mlpBlocks = (nTiles + 7) / 8;          // one tile per wave, 8 waves/block
    if (mlpBlocks > 1024) mlpBlocks = 1024;    // grid-stride handles the rest
    mlp_wmma_kernel<<<mlpBlocks, 256, 0, stream>>>(X, W1, B1, W2, h2, nTiles);
    // 5) out = b2 + self-loop message
    long long total = (long long)N_ * OUT_C;
    init_out_kernel<<<(int)((total + 255) / 256), 256, 0, stream>>>(h2, deg, B2, out, N_);
    // 6) edge gather/scale/scatter-add
    long long sthreads = (long long)E_ * 16;
    scatter_kernel<<<(int)((sthreads + 255) / 256), 256, 0, stream>>>(src, dst, deg, h2, out, E_);
}
